// MoELayer_14199161880838
// MI455X (gfx1250) — matrix operand order repaired and checked by reading, not compiled, not run on hardware
//
#include <hip/hip_runtime.h>
#include <math.h>

// Problem constants (match reference)
#define T_TOK 4096      // B*S
#define DIM   768
#define NEXP  8
#define FFN   3072
#define TM    32        // tokens per FFN tile (2 WMMA M-tiles)
#define T256  256       // threads for helper kernels
#define TFFN  512       // 16 wave32 per FFN workgroup

// LDS strides (bf16 elements, multiples of 8 for 16B-aligned v8bf loads)
#define XS  776         // 768 + 8 pad
#define HS  264         // 256 + 8 pad

typedef __attribute__((ext_vector_type(16))) __bf16 v16bf;
typedef __attribute__((ext_vector_type(8)))  __bf16 v8bf;
typedef __attribute__((ext_vector_type(8)))  float  v8f;

__device__ __forceinline__ __bf16 f2bf(float f) { return (__bf16)f; }

// A-matrix fragment (16x32 bf16): lane<16 holds K {0..7,16..23}, lane>=16 holds
// K {8..15,24..31}. p points at row*stride + kbase + ((lane>=16)?8:0).
__device__ __forceinline__ v16bf frag_a(const __bf16* p) {
  union { v16bf v; v8bf h[2]; } u;
  u.h[0] = *(const v8bf*)(p);
  u.h[1] = *(const v8bf*)(p + 16);
  return u.v;
}

// B-matrix fragment (32x16 bf16): lane<16 holds K {0..7,16..23}, lane>=16 holds
// K {8..15,24..31}. p points at col*stride + kbase + ((lane>=16)?8:0).
__device__ __forceinline__ v16bf frag_b(const __bf16* p) {
  union { v16bf v; v8bf h[2]; } u;
  u.h[0] = *(const v8bf*)(p);
  u.h[1] = *(const v8bf*)(p + 16);
  return u.v;
}

// ---------------------------------------------------------------------------
// Kernel 0: one-time fp32 -> bf16 transpose of expert weights into d_ws.
// src: [E][R][C] fp32 row-major; dst: [E][C][R] bf16 row-major.
// ---------------------------------------------------------------------------
__global__ __launch_bounds__(T256) void transpose_bf16_kernel(
    const float* __restrict__ src, __bf16* __restrict__ dst, int R, int C)
{
  __shared__ __bf16 tile[64 * 72];             // 9 KB, padded
  const int nTr = R >> 6, nTc = C >> 6;
  const int perE = nTr * nTc;
  const int e   = blockIdx.x / perE;
  const int rem = blockIdx.x % perE;
  const int r0  = (rem / nTc) * 64;
  const int c0  = (rem % nTc) * 64;
  const float* s = src + (size_t)e * R * C;
  __bf16*      d = dst + (size_t)e * R * C;

  for (int i = 0; i < 16; ++i) {               // read [r][c], c fastest
    int lin = i * T256 + threadIdx.x;          // 0..4095
    int rr = lin >> 6, cc = lin & 63;
    tile[cc * 72 + rr] = f2bf(s[(size_t)(r0 + rr) * C + (c0 + cc)]);
  }
  __syncthreads();
  for (int i = 0; i < 16; ++i) {               // write [c][r], r fastest
    int lin = i * T256 + threadIdx.x;
    int cc = lin >> 6, rr = lin & 63;
    d[(size_t)(c0 + cc) * R + (r0 + rr)] = tile[cc * 72 + rr];
  }
}

// ---------------------------------------------------------------------------
// Kernel 1: gating — logits, softmax, argmax, p_sel, per-expert token lists
// ---------------------------------------------------------------------------
__global__ __launch_bounds__(T256) void gate_kernel(
    const float* __restrict__ x, const float* __restrict__ Wg,
    float* __restrict__ prob, float* __restrict__ p_sel,
    int* __restrict__ counts, int* __restrict__ idx_list)
{
  __shared__ float wg_s[DIM * NEXP];           // 24 KB
  const int tid = threadIdx.x;
  for (int i = tid; i < DIM * NEXP; i += T256) wg_s[i] = Wg[i];
  __syncthreads();

  const int t = blockIdx.x * T256 + tid;
  if (t >= T_TOK) return;

  const float* xr = x + (size_t)t * DIM;
  float lg[NEXP];
#pragma unroll
  for (int e = 0; e < NEXP; ++e) lg[e] = 0.f;
  for (int d = 0; d < DIM; ++d) {
    float xv = xr[d];
#pragma unroll
    for (int e = 0; e < NEXP; ++e) lg[e] = fmaf(xv, wg_s[d * NEXP + e], lg[e]);
  }

  float m = lg[0]; int am = 0;
#pragma unroll
  for (int e = 1; e < NEXP; ++e) { if (lg[e] > m) { m = lg[e]; am = e; } }
  float s = 0.f; float pe[NEXP];
#pragma unroll
  for (int e = 0; e < NEXP; ++e) { pe[e] = __expf(lg[e] - m); s += pe[e]; }
  float inv = 1.f / s;
#pragma unroll
  for (int e = 0; e < NEXP; ++e) prob[(size_t)t * NEXP + e] = pe[e] * inv;

  p_sel[t] = pe[am] * inv;
  int pos = atomicAdd(&counts[am], 1);
  idx_list[am * T_TOK + pos] = t;
}

// ---------------------------------------------------------------------------
// Kernel 2: balance loss + gate_load tail (deterministic reduction)
// ---------------------------------------------------------------------------
__global__ __launch_bounds__(T256) void loss_kernel(
    const float* __restrict__ prob, const int* __restrict__ counts,
    float* __restrict__ out_tail)
{
  __shared__ float sm[T256];
  const int tid = threadIdx.x;
  float ps[NEXP];
#pragma unroll
  for (int e = 0; e < NEXP; ++e) ps[e] = 0.f;
  for (int t = tid; t < T_TOK; t += T256) {
#pragma unroll
    for (int e = 0; e < NEXP; ++e) ps[e] += prob[(size_t)t * NEXP + e];
  }
  float loss = 0.f;
  for (int e = 0; e < NEXP; ++e) {
    sm[tid] = ps[e];
    __syncthreads();
    for (int s2 = T256 / 2; s2 > 0; s2 >>= 1) {
      if (tid < s2) sm[tid] += sm[tid + s2];
      __syncthreads();
    }
    if (tid == 0)
      loss += (sm[0] / (float)T_TOK) * ((float)counts[e] / (float)T_TOK);
    __syncthreads();
  }
  if (tid == 0) {
    out_tail[0] = (float)NEXP * loss;
    for (int e = 0; e < NEXP; ++e) out_tail[1 + e] = (float)counts[e];
  }
}

// ---------------------------------------------------------------------------
// Kernel 3: routed FFN via bf16 WMMA. One workgroup = (expert, 32-token tile),
// 16 waves. Each wave owns 2 M-tiles that SHARE every B fragment (2 WMMAs per
// B load -> half the L2 weight traffic per FLOP).
// ---------------------------------------------------------------------------
__global__ __launch_bounds__(TFFN) void ffn_kernel(
    const float* __restrict__ x,
    const __bf16* __restrict__ w1t,   // [E][F][D] bf16 (W1 transposed)
    const float* __restrict__ b1,
    const __bf16* __restrict__ w2t,   // [E][D][F] bf16 (W2 transposed)
    const float* __restrict__ b2,
    const float* __restrict__ p_sel,
    const int* __restrict__ counts, const int* __restrict__ idx_list,
    float* __restrict__ out)
{
  const int e    = blockIdx.x >> 7;            // 128 tiles per expert
  const int tile = blockIdx.x & 127;
  const int cnt  = counts[e];
  if (tile * TM >= cnt) return;

  __shared__ __align__(16) __bf16 X_lds[TM * XS];   // 49.7 KB x tile (bf16)
  __shared__ __align__(16) __bf16 h_lds[TM * HS];   // 16.9 KB gelu(h) chunk
  __shared__ int   tok_s[TM];
  __shared__ float ps_s[TM];

  const int tid   = threadIdx.x;
  const int lane  = tid & 31;
  const int wv    = tid >> 5;                  // 0..15
  const int lrow  = lane & 15;                 // A row / B col within tile
  const int kselA = (lane >= 16) ? 8  : 0;     // A fragment K offset
  const int kselB = (lane >= 16) ? 8  : 0;     // B fragment K offset
  const int mhi   = (lane >= 16) ? 8  : 0;     // C row offset

  if (tid < TM) {
    int gi = tile * TM + tid;
    int tk = (gi < cnt) ? idx_list[e * T_TOK + gi] : -1;
    tok_s[tid] = tk;
    ps_s[tid]  = (tk >= 0) ? p_sel[tk] : 0.f;
  }
  __syncthreads();

  // Stage X tile -> bf16 LDS, row-major [32][768]
  for (int r = 0; r < TM; ++r) {
    int tk = tok_s[r];
    const float* xr = x + (size_t)(tk < 0 ? 0 : tk) * DIM;
    for (int c = tid; c < DIM; c += TFFN)
      X_lds[r * XS + c] = f2bf(tk >= 0 ? xr[c] : 0.f);
  }
  __syncthreads();

  // 3 d-tiles x 2 M-tiles of f32 accumulators per wave (48 VGPRs)
  v8f acc0[3], acc1[3];
#pragma unroll
  for (int j = 0; j < 3; ++j) { v8f z = {}; acc0[j] = z; acc1[j] = z; }

  const __bf16* w1e = w1t + (size_t)e * DIM * FFN;   // rows = f, cols = d(K)
  const __bf16* w2e = w2t + (size_t)e * FFN * DIM;   // rows = d, cols = f(K)

  for (int fc = 0; fc < FFN / 256; ++fc) {     // 12 f-chunks of 256
    const int fbase = fc * 256;
    const int fg    = fbase + wv * 16 + lrow;  // this lane's f column

    // ---- GEMM1: h[32 x 256] = X[32 x 768] * W1[768 x 256], K = 768 ----
    v8f hacc0 = {}, hacc1 = {};
    {
      const __bf16* bcol = w1e + (size_t)fg * DIM + kselB;
#pragma unroll 8
      for (int kb = 0; kb < DIM; kb += 32) {
        v16bf b  = frag_b(bcol + kb);                       // shared B
        v16bf a0 = frag_a(&X_lds[lrow * XS + kb + kselA]);
        v16bf a1 = frag_a(&X_lds[(16 + lrow) * XS + kb + kselA]);
        hacc0 = __builtin_amdgcn_wmma_f32_16x16x32_bf16(
            false, a0, false, b, (short)0, hacc0, false, false);
        hacc1 = __builtin_amdgcn_wmma_f32_16x16x32_bf16(
            false, a1, false, b, (short)0, hacc1, false, false);
      }
    }

    // ---- bias + exact GELU -> h_lds (bf16, [m][f_local]) ----
    {
      const float bv = b1[e * FFN + fg];
      __syncthreads();                         // prior h_lds readers done
#pragma unroll
      for (int r = 0; r < 8; ++r) {
        float v0 = hacc0[r] + bv;
        v0 = 0.5f * v0 * (1.f + erff(v0 * 0.70710678118654752f));
        h_lds[(r + mhi) * HS + wv * 16 + lrow] = f2bf(v0);
        float v1 = hacc1[r] + bv;
        v1 = 0.5f * v1 * (1.f + erff(v1 * 0.70710678118654752f));
        h_lds[(16 + r + mhi) * HS + wv * 16 + lrow] = f2bf(v1);
      }
      __syncthreads();                         // h visible to all waves
    }

    // ---- GEMM2: out[32 x 768] += h[32 x 256] * W2[256 x 768] ----
#pragma unroll
    for (int dc = 0; dc < 3; ++dc) {
      const __bf16* bcol =
          w2e + (size_t)(dc * 256 + wv * 16 + lrow) * FFN + fbase + kselB;
#pragma unroll
      for (int kk = 0; kk < 8; ++kk) {
        v16bf b  = frag_b(bcol + kk * 32);                  // shared B
        v16bf a0 = frag_a(&h_lds[lrow * HS + kk * 32 + kselA]);
        v16bf a1 = frag_a(&h_lds[(16 + lrow) * HS + kk * 32 + kselA]);
        acc0[dc] = __builtin_amdgcn_wmma_f32_16x16x32_bf16(
            false, a0, false, b, (short)0, acc0[dc], false, false);
        acc1[dc] = __builtin_amdgcn_wmma_f32_16x16x32_bf16(
            false, a1, false, b, (short)0, acc1[dc], false, false);
      }
    }
  }

  // ---- epilogue: out[tok, d] = p_sel * (acc + b2[d]) ----
#pragma unroll
  for (int j = 0; j < 3; ++j) {
    const int dg = j * 256 + wv * 16 + lrow;
    const float bb = b2[e * DIM + dg];
#pragma unroll
    for (int r = 0; r < 8; ++r) {
      int m0 = r + mhi;
      int tk0 = tok_s[m0];
      if (tk0 >= 0)
        out[(size_t)tk0 * DIM + dg] = ps_s[m0] * (acc0[j][r] + bb);
      int m1 = 16 + r + mhi;
      int tk1 = tok_s[m1];
      if (tk1 >= 0)
        out[(size_t)tk1 * DIM + dg] = ps_s[m1] * (acc1[j][r] + bb);
    }
  }
}

// ---------------------------------------------------------------------------
extern "C" void kernel_launch(void* const* d_in, const int* in_sizes, int n_in,
                              void* d_out, int out_size, void* d_ws, size_t ws_size,
                              hipStream_t stream) {
  const float* x  = (const float*)d_in[0];
  // d_in[1] input_ids, d_in[2] attention_mask: unused by the math
  const float* Wg = (const float*)d_in[3];
  const float* W1 = (const float*)d_in[4];
  const float* b1 = (const float*)d_in[5];
  const float* W2 = (const float*)d_in[6];
  const float* b2 = (const float*)d_in[7];
  float* out = (float*)d_out;

  // workspace carve-up
  float* prob = (float*)d_ws;                        // [T, E]
  float* psel = prob + (size_t)T_TOK * NEXP;         // [T]
  int*   cnts = (int*)(psel + T_TOK);                // [E]
  int*   idxl = cnts + NEXP;                         // [E, T]
  uintptr_t wp = (uintptr_t)(idxl + (size_t)NEXP * T_TOK);
  wp = (wp + 255u) & ~(uintptr_t)255u;               // align bf16 cache
  __bf16* w1t = (__bf16*)wp;                         // [E][F][D] 37.75 MB
  __bf16* w2t = w1t + (size_t)NEXP * DIM * FFN;      // [E][D][F] 37.75 MB

  hipMemsetAsync(cnts, 0, NEXP * sizeof(int), stream);

  // one-time weight convert+transpose into L2-resident bf16 cache
  const int tpb = NEXP * (DIM / 64) * (FFN / 64);    // 4608 tiles each
  transpose_bf16_kernel<<<tpb, T256, 0, stream>>>(W1, w1t, DIM, FFN);
  transpose_bf16_kernel<<<tpb, T256, 0, stream>>>(W2, w2t, FFN, DIM);

  gate_kernel<<<T_TOK / T256, T256, 0, stream>>>(
      x, Wg, prob, psel, cnts, idxl);

  loss_kernel<<<1, T256, 0, stream>>>(
      prob, cnts, out + (size_t)T_TOK * DIM);

  // worst case: one expert owns all 4096 tokens -> 128 tiles per expert
  ffn_kernel<<<NEXP * (T_TOK / TM), TFFN, 0, stream>>>(
      x, w1t, b1, w2t, b2, psel, cnts, idxl, out);
}
